// WeightOnlyPerChannelQuantizedLinear_88708254532293
// MI455X (gfx1250) — compile-verified, matrix operand order checked by reading, not run on hardware
//
#include <hip/hip_runtime.h>
#include <stdint.h>

typedef _Float16 half4  __attribute__((ext_vector_type(4)));
typedef _Float16 half8  __attribute__((ext_vector_type(8)));
typedef _Float16 half16 __attribute__((ext_vector_type(16)));
typedef float    f32x4  __attribute__((ext_vector_type(4)));
typedef float    f32x8  __attribute__((ext_vector_type(8)));

// exact parameter type from the hipcc diagnostic: vector_size(16) int
typedef int v4i_vs __attribute__((vector_size(16)));
typedef __attribute__((address_space(1))) v4i_vs* g_v4i_ptr;
typedef __attribute__((address_space(3))) v4i_vs* l_v4i_ptr;

#define MDIM 8192   // B*S
#define NDIM 4096   // D_OUT
#define KDIM 4096   // D_IN

#define BM 128
#define BN 128

// ---------------- CDNA5 async global->LDS detection ----------------
#if defined(__has_builtin)
#  if __has_builtin(__builtin_amdgcn_global_load_async_to_lds_b128) && \
      __has_builtin(__builtin_amdgcn_s_wait_asynccnt)
#    define USE_ASYNC_LDS 1
#  endif
#endif
#ifndef USE_ASYNC_LDS
#  define USE_ASYNC_LDS 0
#endif

__device__ __forceinline__ void copy16_g2lds(const void* g, void* l) {
#if USE_ASYNC_LDS
    // flat->AS1 is identity; flat LDS addr[31:0] is the wave-relative LDS offset (ISA 10.2)
    __builtin_amdgcn_global_load_async_to_lds_b128(
        (g_v4i_ptr)(uintptr_t)g,
        (l_v4i_ptr)(uint32_t)(uintptr_t)l,
        0, 0);
#else
    *(half8*)l = *(const half8*)g;
#endif
}

__device__ __forceinline__ void wait_async_copies() {
#if USE_ASYNC_LDS
    __builtin_amdgcn_s_wait_asynccnt(0);
#endif
}

// ---------------- one-shot dtype conversion kernels ----------------
// X: fp32 -> f16 (8 elems / thread)
__launch_bounds__(256)
__global__ void cvt_x_f16_kernel(const float* __restrict__ x, _Float16* __restrict__ xh) {
    const size_t i = ((size_t)blockIdx.x * 256 + threadIdx.x) * 8;
    f32x4 a = *(const f32x4*)(x + i);
    f32x4 b = *(const f32x4*)(x + i + 4);
    half8 h;
#pragma unroll
    for (int j = 0; j < 4; ++j) { h[j] = (_Float16)a[j]; h[j + 4] = (_Float16)b[j]; }
    *(half8*)(xh + i) = h;
}

// W: int8 -> f16 (exact; 16 elems / thread)
__launch_bounds__(256)
__global__ void cvt_w_f16_kernel(const signed char* __restrict__ w, _Float16* __restrict__ wh) {
    const size_t i = ((size_t)blockIdx.x * 256 + threadIdx.x) * 16;
    int4 v = *(const int4*)(w + i);
    const signed char* pb = (const signed char*)&v;
    half8 h0, h1;
#pragma unroll
    for (int j = 0; j < 8; ++j) { h0[j] = (_Float16)pb[j]; h1[j] = (_Float16)pb[j + 8]; }
    *(half8*)(wh + i)     = h0;
    *(half8*)(wh + i + 8) = h1;
}

// ---------------- main GEMM on pre-converted f16 (async staged) ----------------
#define BK2  64
#define LDK2 72   // padded row stride in halves (144B: 16B aligned, conflict-light)
#define T2   (KDIM / BK2)

__launch_bounds__(256)
__global__ void wq_gemm_f16_preconv(const _Float16* __restrict__ xh,
                                    const _Float16* __restrict__ wh,
                                    const float* __restrict__ scaler,
                                    float* __restrict__ out)
{
    __shared__ _Float16 lA[2][BM * LDK2];
    __shared__ _Float16 lB[2][BN * LDK2];

    const int tid    = threadIdx.x;
    const int lane   = tid & 31;
    const int wv     = tid >> 5;
    const int wm     = wv & 1;
    const int wn     = wv >> 1;
    const int laneLo = lane & 15;
    const int laneHi = lane >> 4;

    const int m0 = blockIdx.y * BM;
    const int n0 = blockIdx.x * BN;

    f32x8 acc[4][2];
#pragma unroll
    for (int i = 0; i < 4; ++i)
#pragma unroll
        for (int j = 0; j < 2; ++j)
            acc[i][j] = (f32x8)0.0f;

    // tile = 128 rows x 128 bytes -> 1024 x 16B chunks; 4 chunks / thread / matrix
    auto copy_tile = [&](int buf, int kt) {
        const int k0 = kt * BK2;
#pragma unroll
        for (int q = 0; q < 4; ++q) {
            const int id = tid + 256 * q;
            const int r  = id >> 3;
            const int c  = id & 7;
            copy16_g2lds(xh + (size_t)(m0 + r) * KDIM + k0 + c * 8,
                         &lA[buf][r * LDK2 + c * 8]);
            copy16_g2lds(wh + (size_t)(n0 + r) * KDIM + k0 + c * 8,
                         &lB[buf][r * LDK2 + c * 8]);
        }
    };

    auto compute = [&](int buf) {
#pragma unroll
        for (int kk = 0; kk < 2; ++kk) {
            const int koff = kk * 32 + laneHi * 16;
            half16 afrag[4], bfrag[2];
#pragma unroll
            for (int mi = 0; mi < 4; ++mi) {
                const _Float16* p = &lA[buf][(wm * 64 + mi * 16 + laneLo) * LDK2 + koff];
                half8 lo = *(const half8*)p;
                half8 hi = *(const half8*)(p + 8);
                afrag[mi] = __builtin_shufflevector(lo, hi,
                    0, 1, 2, 3, 4, 5, 6, 7, 8, 9, 10, 11, 12, 13, 14, 15);
            }
#pragma unroll
            for (int ni = 0; ni < 2; ++ni) {
                const _Float16* p = &lB[buf][(wn * 32 + ni * 16 + laneLo) * LDK2 + koff];
                half8 lo = *(const half8*)p;
                half8 hi = *(const half8*)(p + 8);
                bfrag[ni] = __builtin_shufflevector(lo, hi,
                    0, 1, 2, 3, 4, 5, 6, 7, 8, 9, 10, 11, 12, 13, 14, 15);
            }
#pragma unroll
            for (int mi = 0; mi < 4; ++mi)
#pragma unroll
                for (int ni = 0; ni < 2; ++ni)
                    acc[mi][ni] = __builtin_amdgcn_wmma_f32_16x16x32_f16(
                        false, afrag[mi], false, bfrag[ni],
                        (short)0, acc[mi][ni], false, false);
        }
    };

    copy_tile(0, 0);
    for (int t = 0; t < T2; ++t) {
        wait_async_copies();               // my ASYNCcnt -> 0 (tile t landed in LDS)
        __syncthreads();                   // everyone's copies landed; prior readers done
        if (t + 1 < T2) copy_tile((t + 1) & 1, t + 1);   // fill other buffer under WMMA
        if (t + 2 < T2) {
            const int k2 = (t + 2) * BK2;
            __builtin_prefetch(xh + (size_t)(m0 + (tid >> 1)) * KDIM + k2, 0, 3);
            __builtin_prefetch(wh + (size_t)(n0 + (tid >> 1)) * KDIM + k2, 0, 3);
        }
        compute(t & 1);
    }

#pragma unroll
    for (int ni = 0; ni < 2; ++ni) {
        const int col = n0 + wn * 32 + ni * 16 + laneLo;
        const float sc = scaler[col];
#pragma unroll
        for (int mi = 0; mi < 4; ++mi) {
            const int rowBase = m0 + wm * 64 + mi * 16 + laneHi * 8;
#pragma unroll
            for (int r = 0; r < 8; ++r)
                out[(size_t)(rowBase + r) * NDIM + col] = acc[mi][ni][r] * sc;
        }
    }
}

// ---------------- fallback: fused convert-in-kernel GEMM (no workspace) ----------------
#define BK  32
#define LDK 40

__launch_bounds__(256)
__global__ void wq_gemm_fused(const float* __restrict__ x,
                              const signed char* __restrict__ w,
                              const float* __restrict__ scaler,
                              float* __restrict__ out)
{
    __shared__ _Float16 lA[2][BM * LDK];
    __shared__ _Float16 lB[2][BN * LDK];

    const int tid    = threadIdx.x;
    const int lane   = tid & 31;
    const int wv     = tid >> 5;
    const int wm     = wv & 1;
    const int wn     = wv >> 1;
    const int laneLo = lane & 15;
    const int laneHi = lane >> 4;

    const int m0 = blockIdx.y * BM;
    const int n0 = blockIdx.x * BN;

    const int wrow = tid >> 1;
    const int wh   = tid & 1;

    f32x4 xs[4];
    int4  wsv;

    f32x8 acc[4][2];
#pragma unroll
    for (int i = 0; i < 4; ++i)
#pragma unroll
        for (int j = 0; j < 2; ++j)
            acc[i][j] = (f32x8)0.0f;

    const int T = KDIM / BK;

    auto load_tile = [&](int kt) {
        const int k0 = kt * BK;
#pragma unroll
        for (int q = 0; q < 4; ++q) {
            const int id = tid + 256 * q;
            const int r  = id >> 3;
            const int c4 = id & 7;
            xs[q] = *(const f32x4*)(x + (size_t)(m0 + r) * KDIM + k0 + c4 * 4);
        }
        wsv = *(const int4*)(w + (size_t)(n0 + wrow) * KDIM + k0 + wh * 16);
    };

    auto store_tile = [&](int buf) {
#pragma unroll
        for (int q = 0; q < 4; ++q) {
            const int id = tid + 256 * q;
            const int r  = id >> 3;
            const int c4 = id & 7;
            half4 h;
            h[0] = (_Float16)xs[q][0];
            h[1] = (_Float16)xs[q][1];
            h[2] = (_Float16)xs[q][2];
            h[3] = (_Float16)xs[q][3];
            *(half4*)(&lA[buf][r * LDK + c4 * 4]) = h;
        }
        const signed char* pb = (const signed char*)&wsv;
        half8 b0, b1;
#pragma unroll
        for (int i = 0; i < 8; ++i) { b0[i] = (_Float16)pb[i]; b1[i] = (_Float16)pb[i + 8]; }
        _Float16* dst = &lB[buf][wrow * LDK + wh * 16];
        *(half8*)(dst)     = b0;
        *(half8*)(dst + 8) = b1;
    };

    auto compute = [&](int buf) {
        half16 afrag[4], bfrag[2];
#pragma unroll
        for (int mi = 0; mi < 4; ++mi) {
            const _Float16* p = &lA[buf][(wm * 64 + mi * 16 + laneLo) * LDK + laneHi * 16];
            half8 lo = *(const half8*)p;
            half8 hi = *(const half8*)(p + 8);
            afrag[mi] = __builtin_shufflevector(lo, hi,
                0, 1, 2, 3, 4, 5, 6, 7, 8, 9, 10, 11, 12, 13, 14, 15);
        }
#pragma unroll
        for (int ni = 0; ni < 2; ++ni) {
            const _Float16* p = &lB[buf][(wn * 32 + ni * 16 + laneLo) * LDK + laneHi * 16];
            half8 lo = *(const half8*)p;
            half8 hi = *(const half8*)(p + 8);
            bfrag[ni] = __builtin_shufflevector(lo, hi,
                0, 1, 2, 3, 4, 5, 6, 7, 8, 9, 10, 11, 12, 13, 14, 15);
        }
#pragma unroll
        for (int mi = 0; mi < 4; ++mi)
#pragma unroll
            for (int ni = 0; ni < 2; ++ni)
                acc[mi][ni] = __builtin_amdgcn_wmma_f32_16x16x32_f16(
                    false, afrag[mi], false, bfrag[ni],
                    (short)0, acc[mi][ni], false, false);
    };

    load_tile(0);
    for (int t = 0; t < T; ++t) {
        __syncthreads();
        store_tile(t & 1);
        if (t + 1 < T) load_tile(t + 1);
        if (t + 2 < T) {
            const int k2 = (t + 2) * BK;
            __builtin_prefetch(x + (size_t)(m0 + (tid >> 1)) * KDIM + k2, 0, 3);
            __builtin_prefetch(w + (size_t)(n0 + wrow) * KDIM + k2, 0, 3);
        }
        __syncthreads();
        compute(t & 1);
    }

#pragma unroll
    for (int ni = 0; ni < 2; ++ni) {
        const int col = n0 + wn * 32 + ni * 16 + laneLo;
        const float sc = scaler[col];
#pragma unroll
        for (int mi = 0; mi < 4; ++mi) {
            const int rowBase = m0 + wm * 64 + mi * 16 + laneHi * 8;
#pragma unroll
            for (int r = 0; r < 8; ++r)
                out[(size_t)(rowBase + r) * NDIM + col] = acc[mi][ni][r] * sc;
        }
    }
}

// ---------------- launch ----------------
extern "C" void kernel_launch(void* const* d_in, const int* in_sizes, int n_in,
                              void* d_out, int out_size, void* d_ws, size_t ws_size,
                              hipStream_t stream)
{
    const float*       x  = (const float*)d_in[0];
    const signed char* w  = (const signed char*)d_in[1];
    const float*       sc = (const float*)d_in[2];
    float*             out = (float*)d_out;

    const size_t xh_bytes = (size_t)MDIM * KDIM * sizeof(_Float16); // 64 MiB
    const size_t wh_bytes = (size_t)NDIM * KDIM * sizeof(_Float16); // 32 MiB

    dim3 grid(NDIM / BN, MDIM / BM);   // (32, 64)
    dim3 block(256);

    if (ws_size >= xh_bytes + wh_bytes) {
        _Float16* xh = (_Float16*)d_ws;
        _Float16* wh = (_Float16*)((char*)d_ws + xh_bytes);
        cvt_x_f16_kernel<<<(MDIM * (size_t)KDIM) / (256 * 8),  256, 0, stream>>>(x, xh);
        cvt_w_f16_kernel<<<((size_t)NDIM * KDIM) / (256 * 16), 256, 0, stream>>>(w, wh);
        wq_gemm_f16_preconv<<<grid, block, 0, stream>>>(xh, wh, sc, out);
    } else {
        wq_gemm_fused<<<grid, block, 0, stream>>>(x, w, sc, out);
    }
}
